// RealAgnosticResidualNonLinearInteractionBlock_84129819394066
// MI455X (gfx1250) — matrix operand order repaired and checked by reading, not compiled, not run on hardware
//
#include <hip/hip_runtime.h>
#include <hip/hip_bf16.h>

// ---------------------------------------------------------------------------
// MACE-style interaction block for MI455X (gfx1250), wave32 + WMMA f16->f32.
// All GEMMs use v_wmma_f32_16x16x32_f16.
//  * B weights pre-swizzled to fragment order (1 contiguous 32B read/lane).
//  * A tiles staged in LDS in fragment order (1 contiguous 32B LDS read/lane).
//  * Wave-uniform task decode via readfirstlane -> scalar control flow.
//  * Edge pipeline fully fused in LDS; scatter via f32 atomics (L2-resident).
// Assumes N % 16 == 0 and E % 16 == 0 (true here: 10000 / 160000).
// ---------------------------------------------------------------------------

typedef __attribute__((ext_vector_type(16))) _Float16 v16h;
typedef __attribute__((ext_vector_type(8)))  float    v8f;

#define INV_SQRT3 0.5773502691896258f

__device__ __forceinline__ v8f wmma16(v16h a, v16h b, v8f c) {
  // (neg_a, A, neg_b, B, c_mod, C, reuse_a, reuse_b)
  return __builtin_amdgcn_wmma_f32_16x16x32_f16(false, a, false, b, (short)0, c,
                                                false, false);
}

__device__ __forceinline__ float siluf(float x) { return x / (1.0f + __expf(-x)); }
__device__ __forceinline__ float sigmf(float x) { return 1.0f / (1.0f + __expf(-x)); }

// Swizzled A-tile index: element (row m, col k) of a 16xK tile stored so that
// lane l of k-tile kt reads 16 contiguous halves (ISA 16-bit A 16x32 layout).
__device__ __forceinline__ int a_sw_idx(int m, int k) {
  int kt = k >> 5;
  int kk = k & 31;
  int g  = kk >> 3;                       // 8-wide K group
  int lane = m + ((g & 1) << 4);
  int t    = (kk & 7) + ((g >> 1) << 3);
  return (kt << 9) + (lane << 4) + t;
}

// A fragment from swizzled LDS: one 32B contiguous read per lane.
__device__ __forceinline__ v16h load_a_sw(const _Float16* A, int kt, int lane) {
  return *(const v16h*)(A + (kt << 9) + (lane << 4));
}

// A fragment from global f32 (exact K, no guards), f32->f16 on the fly.
__device__ __forceinline__ v16h load_a_g_nc(const float* A, int rowStride,
                                            int elemStride, int kt, int lane) {
  int m = lane & 15;
  int koff = (lane >> 4) << 3;
  const float* rp = A + (size_t)m * rowStride + (size_t)kt * 32 * elemStride;
  v16h r;
#pragma unroll
  for (int j = 0; j < 8; ++j) {
    int kk = ((j < 4) ? 0 : 16) + koff + 2 * (j & 3);
    r[2 * j]     = (_Float16)rp[(size_t)kk * elemStride];
    r[2 * j + 1] = (_Float16)rp[(size_t)(kk + 1) * elemStride];
  }
  return r;
}

// B fragment from pre-swizzled weights (1 contiguous 32B read per lane).
__device__ __forceinline__ v16h load_b_sw(const _Float16* B, int ntcount, int kt,
                                          int nt, int lane) {
  return *(const v16h*)(B + (((size_t)(kt * ntcount + nt)) << 9) + (lane << 4));
}

// ---------------------------------------------------------------------------
// Weight conversion + B-fragment swizzle. mode 0: identity rows, zero pad
// beyond Ksrc. mode 1: radial-MLP input padding map (rows 0..7 = edge_feats,
// 8..31 zero, 32..159 = src rows 8..135, 160..287 = src rows 136..263).
// ---------------------------------------------------------------------------
__global__ void k_swizzleB(const float* __restrict__ src, _Float16* __restrict__ dst,
                           int Ksrc, int Nw, int ktiles, int mode) {
  int total = ktiles * (Nw / 16) * 512;
  for (int idx = blockIdx.x * blockDim.x + threadIdx.x; idx < total;
       idx += gridDim.x * blockDim.x) {
    int f    = idx >> 9;
    int r    = idx & 511;
    int lane = r >> 4;
    int t    = r & 15;
    int ntc  = Nw / 16;
    int nt   = f % ntc;
    int kt   = f / ntc;
    int krow = kt * 32 + ((t >> 3) << 4) + ((lane >> 4) << 3) + (t & 7);
    int ncol = nt * 16 + (lane & 15);
    int srow;
    if (mode == 1) {
      if (krow < 8)       srow = krow;
      else if (krow < 32) srow = -1;
      else                srow = krow - 24;
    } else {
      srow = krow;
    }
    float v = (srow >= 0 && srow < Ksrc) ? src[(size_t)srow * Nw + ncol] : 0.0f;
    dst[idx] = (_Float16)v;
  }
}

// ---------------------------------------------------------------------------
// Node-side pre-GEMMs: sc_s/sc_v (-> d_out), up_s/up_v, res_s/res_v,
// src_emb/tgt_emb (f16). 16 nodes per block, 8 waves, 120 WMMA n-tile tasks.
// A tiles staged once in swizzled f16 LDS (~17 KB).
// ---------------------------------------------------------------------------
__global__ void __launch_bounds__(256)
k_node_pre(const float* __restrict__ node_feats, const float* __restrict__ node_attrs,
           const _Float16* __restrict__ w_skip0, const _Float16* __restrict__ w_up0,
           const _Float16* __restrict__ w_res0, const _Float16* __restrict__ w_skip1,
           const _Float16* __restrict__ w_up1, const _Float16* __restrict__ w_res1,
           const _Float16* __restrict__ w_srcp, const _Float16* __restrict__ w_tgtp,
           float* __restrict__ out_sc, float* __restrict__ up_s,
           float* __restrict__ up_v, float* __restrict__ res_s,
           float* __restrict__ res_v, _Float16* __restrict__ src_emb16,
           _Float16* __restrict__ tgt_emb16, int N) {
  __shared__ __attribute__((aligned(32))) _Float16 sA[4 * 512];
  __shared__ __attribute__((aligned(32))) _Float16 vA[3][4 * 512];
  __shared__ __attribute__((aligned(32))) _Float16 atA[512];

  int nb   = blockIdx.x * 16;
  int tid  = threadIdx.x;
  int wave = __builtin_amdgcn_readfirstlane(tid >> 5);  // wave-uniform -> SGPR
  int lane = tid & 31;

  // Stage A tiles (swizzled f16): s 16x128, v[3] 16x128, attrs 16x32 (pad K).
  for (int idx = tid; idx < 2048; idx += 256) {
    int m = idx >> 7, k = idx & 127;
    sA[a_sw_idx(m, k)] = (_Float16)node_feats[(size_t)(nb + m) * 512 + k];
  }
  for (int idx = tid; idx < 6144; idx += 256) {
    int comp = idx >> 11, r = idx & 2047;
    int m = r >> 7, k = r & 127;
    vA[comp][a_sw_idx(m, k)] =
        (_Float16)node_feats[(size_t)(nb + m) * 512 + 128 + k * 3 + comp];
  }
  for (int idx = tid; idx < 512; idx += 256) {
    int m = idx >> 5, k = idx & 31;
    atA[a_sw_idx(m, k)] =
        (k < 10) ? (_Float16)node_attrs[(size_t)(nb + m) * 10 + k] : (_Float16)0.0f;
  }
  __syncthreads();

  for (int task = wave; task < 120; task += 8) {
    // ---- scalar task decode (all operands wave-uniform) ----
    const _Float16* A;
    const _Float16* B;
    float*    fbase = nullptr;   // f32 destination descriptor
    _Float16* hbase = nullptr;   // f16 destination (embeddings)
    int rstride = 0, cmul = 1, coff = 0;
    int ntile, ktiles, ntc;
    if (task < 32) {                       // scalar channel GEMMs, K=128
      A = sA; ktiles = 4;
      if (task < 8)       { B = w_skip0; ntc = 8;  ntile = task;      fbase = out_sc; rstride = 512; }
      else if (task < 16) { B = w_up0;   ntc = 8;  ntile = task - 8;  fbase = up_s;   rstride = 128; }
      else                { B = w_res0;  ntc = 16; ntile = task - 16; fbase = res_s;  rstride = 256; }
    } else if (task < 104) {               // vector channel GEMMs x3 comps
      int t = task - 32; int comp = t / 24; int r = t % 24;
      A = vA[comp]; ktiles = 4; ntc = 8; cmul = 3;
      if (r < 8)       { B = w_skip1; ntile = r;      fbase = out_sc; rstride = 512; coff = 128 + comp; }
      else if (r < 16) { B = w_up1;   ntile = r - 8;  fbase = up_v;   rstride = 384; coff = comp; }
      else             { B = w_res1;  ntile = r - 16; fbase = res_v;  rstride = 384; coff = comp; }
    } else {                               // attribute embeddings, K=10 (pad 32)
      int t = task - 104; A = atA; ktiles = 1; ntc = 8;
      if (t < 8) { B = w_srcp; ntile = t;     hbase = src_emb16; }
      else       { B = w_tgtp; ntile = t - 8; hbase = tgt_emb16; }
    }

    v8f acc = {};
    for (int kt = 0; kt < ktiles; ++kt)
      acc = wmma16(load_a_sw(A, kt, lane), load_b_sw(B, ntc, kt, ntile, lane), acc);

    int col   = ntile * 16 + (lane & 15);
    int mbase = (lane >> 4) * 8;
    if (hbase == nullptr) {
      float* p = fbase + (size_t)(nb + mbase) * rstride + col * cmul + coff;
#pragma unroll
      for (int j = 0; j < 8; ++j) p[(size_t)j * rstride] = acc[j];
    } else {
      _Float16* p = hbase + (size_t)(nb + mbase) * 128 + col;
#pragma unroll
      for (int j = 0; j < 8; ++j) p[(size_t)j * 128] = (_Float16)acc[j];
    }
  }
}

// ---------------------------------------------------------------------------
// Fused edge kernel: 16 edges per block, 4 waves. aug assembly (swizzled) ->
// 4-layer WMMA MLP + density head in LDS -> tensor-product msgs -> atomics.
// ---------------------------------------------------------------------------
__global__ void __launch_bounds__(128)
k_edge(const float* __restrict__ edge_feats, const float* __restrict__ edge_attrs,
       const int* __restrict__ edge_index, const _Float16* __restrict__ src_emb16,
       const _Float16* __restrict__ tgt_emb16, const _Float16* __restrict__ w_rad1p,
       const _Float16* __restrict__ w_rad2, const _Float16* __restrict__ w_rad3,
       const _Float16* __restrict__ w_rad4, const _Float16* __restrict__ w_den1p,
       const float* __restrict__ W_den2, const float* __restrict__ up_s,
       const float* __restrict__ up_v, float* __restrict__ msg_s,
       float* __restrict__ msg_v, float* __restrict__ density, int E) {
  __shared__ __attribute__((aligned(32))) _Float16 augL[9 * 512];   // 16x288 swz
  __shared__ __attribute__((aligned(32))) _Float16 h1L[2 * 512];    // 16x64 swz
  __shared__ __attribute__((aligned(32))) _Float16 hdenL[2 * 512];
  __shared__ __attribute__((aligned(32))) _Float16 h2L[2 * 512];
  __shared__ __attribute__((aligned(32))) _Float16 h3L[2 * 512];
  __shared__ float tpwL[16 * 512];
  __shared__ int   sndL[16], rcvL[16];

  int ge0  = blockIdx.x * 16;
  int tid  = threadIdx.x;
  int wave = __builtin_amdgcn_readfirstlane(tid >> 5);  // wave-uniform -> SGPR
  int lane = tid & 31;

  if (tid < 16) {
    sndL[tid] = edge_index[(size_t)(ge0 + tid) * 2 + 0];
    rcvL[tid] = edge_index[(size_t)(ge0 + tid) * 2 + 1];
  }
  // Build aug (16 x 288 f16, swizzled): [0:8)=edge_feats, [8:32)=0,
  // [32:160)=src_emb, [160:288)=tgt_emb. Matches padded weight row map.
  // e is loop-uniform -> edge_index loads scalarize; k strided by blockDim.
  for (int e = 0; e < 16; ++e) {
    int ge  = ge0 + e;
    int snd = edge_index[(size_t)ge * 2 + 0];
    int rcv = edge_index[(size_t)ge * 2 + 1];
    for (int k = tid; k < 288; k += 128) {
      _Float16 v;
      if (k < 8)        v = (_Float16)edge_feats[(size_t)ge * 8 + k];
      else if (k < 32)  v = (_Float16)0.0f;
      else if (k < 160) v = src_emb16[(size_t)snd * 128 + (k - 32)];
      else              v = tgt_emb16[(size_t)rcv * 128 + (k - 160)];
      augL[a_sw_idx(e, k)] = v;
    }
  }
  __syncthreads();

  { // layer 1 (radial) + layer 1 (density): K=288 (9 k-tiles), N=64 each
    v8f accR = {}, accD = {};
    for (int kt = 0; kt < 9; ++kt) {
      v16h a = load_a_sw(augL, kt, lane);
      if (kt < 8) __builtin_prefetch(w_rad1p + (((kt + 1) * 4 + wave) << 9), 0, 0);
      accR = wmma16(a, load_b_sw(w_rad1p, 4, kt, wave, lane), accR);
      accD = wmma16(a, load_b_sw(w_den1p, 4, kt, wave, lane), accD);
    }
    int n = wave * 16 + (lane & 15);
    int mbase = (lane >> 4) * 8;
#pragma unroll
    for (int j = 0; j < 8; ++j) {
      h1L[a_sw_idx(mbase + j, n)]   = (_Float16)siluf(accR[j]);
      hdenL[a_sw_idx(mbase + j, n)] = (_Float16)siluf(accD[j]);
    }
  }
  __syncthreads();

  // density head: edge_density = tanh((silu(aug@Wd1) @ Wd2)^2) -> scatter
  if (tid < 16) {
    float dot = 0.0f;
    for (int k = 0; k < 64; ++k) dot += (float)hdenL[a_sw_idx(tid, k)] * W_den2[k];
    atomicAdd(&density[rcvL[tid]], tanhf(dot * dot));
  }
  { // layer 2
    v8f acc = {};
    for (int kt = 0; kt < 2; ++kt)
      acc = wmma16(load_a_sw(h1L, kt, lane), load_b_sw(w_rad2, 4, kt, wave, lane), acc);
    int n = wave * 16 + (lane & 15);
    int mbase = (lane >> 4) * 8;
#pragma unroll
    for (int j = 0; j < 8; ++j) h2L[a_sw_idx(mbase + j, n)] = (_Float16)siluf(acc[j]);
  }
  __syncthreads();
  { // layer 3
    v8f acc = {};
    for (int kt = 0; kt < 2; ++kt)
      acc = wmma16(load_a_sw(h2L, kt, lane), load_b_sw(w_rad3, 4, kt, wave, lane), acc);
    int n = wave * 16 + (lane & 15);
    int mbase = (lane >> 4) * 8;
#pragma unroll
    for (int j = 0; j < 8; ++j) h3L[a_sw_idx(mbase + j, n)] = (_Float16)siluf(acc[j]);
  }
  __syncthreads();
  // layer 4: K=64, N=512 -> 32 n-tiles, 8 per wave; tpw to LDS f32
  for (int nt = wave; nt < 32; nt += 4) {
    v8f acc = {};
    for (int kt = 0; kt < 2; ++kt)
      acc = wmma16(load_a_sw(h3L, kt, lane), load_b_sw(w_rad4, 32, kt, nt, lane), acc);
    int n = nt * 16 + (lane & 15);
    int mbase = (lane >> 4) * 8;
#pragma unroll
    for (int j = 0; j < 8; ++j) tpwL[(mbase + j) * 512 + n] = acc[j];
  }
  __syncthreads();

  // Tensor-product messages + atomic scatter (targets are L2-resident).
  int u = tid;  // 128 threads <-> 128 channels
  for (int e = 0; e < 16; ++e) {
    int ge  = ge0 + e;
    int snd = sndL[e], rcv = rcvL[e];
    float y0  = edge_attrs[(size_t)ge * 4 + 0];
    float y11 = edge_attrs[(size_t)ge * 4 + 1];
    float y12 = edge_attrs[(size_t)ge * 4 + 2];
    float y13 = edge_attrs[(size_t)ge * 4 + 3];
    float wA = tpwL[e * 512 + u];
    float wB = tpwL[e * 512 + 128 + u];
    float wC = tpwL[e * 512 + 256 + u];
    float wD = tpwL[e * 512 + 384 + u];
    float xs  = up_s[(size_t)snd * 128 + u];
    float xv0 = up_v[(size_t)snd * 384 + u * 3 + 0];
    float xv1 = up_v[(size_t)snd * 384 + u * 3 + 1];
    float xv2 = up_v[(size_t)snd * 384 + u * 3 + 2];
    float mA   = wA * xs * y0;
    float mD   = wD * (xv0 * y11 + xv1 * y12 + xv2 * y13) * INV_SQRT3;
    float wBxs = wB * xs;
    float wCy0 = wC * y0;
    float* ms = msg_s + (size_t)rcv * 256;
    float* mv = msg_v + (size_t)rcv * 768;
    atomicAdd(ms + u, mA);
    atomicAdd(ms + 128 + u, mD);
    atomicAdd(mv + u * 3 + 0, wBxs * y11);
    atomicAdd(mv + u * 3 + 1, wBxs * y12);
    atomicAdd(mv + u * 3 + 2, wBxs * y13);
    atomicAdd(mv + (128 + u) * 3 + 0, wCy0 * xv0);
    atomicAdd(mv + (128 + u) * 3 + 1, wCy0 * xv1);
    atomicAdd(mv + (128 + u) * 3 + 2, wCy0 * xv2);
  }
}

// ---------------------------------------------------------------------------
// Node-side post: lin1 GEMMs -> denom/residual/gate elementwise -> lin2 GEMMs
// -> message written to d_out. 16 nodes per block, 8 waves.
// ---------------------------------------------------------------------------
__global__ void __launch_bounds__(256)
k_node_post(const float* __restrict__ msg_s, const float* __restrict__ msg_v,
            const float* __restrict__ density, const float* __restrict__ res_s,
            const float* __restrict__ res_v, const _Float16* __restrict__ w_lin1_0,
            const _Float16* __restrict__ w_lin1_1, const _Float16* __restrict__ w_lin2_0,
            const _Float16* __restrict__ w_lin2_1, const float* __restrict__ alpha_p,
            const float* __restrict__ beta_p, float* __restrict__ out_msg, int N) {
  __shared__ float t_sL[16 * 256];
  __shared__ float t_vL[16 * 384];
  __shared__ __attribute__((aligned(32))) _Float16 scalL[4 * 512];   // 16x128 swz
  __shared__ __attribute__((aligned(32))) _Float16 vecL[3][4 * 512];

  int nb   = blockIdx.x * 16;
  int tid  = threadIdx.x;
  int wave = __builtin_amdgcn_readfirstlane(tid >> 5);  // wave-uniform -> SGPR
  int lane = tid & 31;

  // lin1: msg_s @ W_lin1_0 (16 tiles) and msg_v @ W_lin1_1 (24 tiles); K=256
  for (int task = wave; task < 40; task += 8) {
    v8f acc = {};
    if (task < 16) {
      for (int kt = 0; kt < 8; ++kt) {
        v16h a = load_a_g_nc(msg_s + (size_t)nb * 256, 256, 1, kt, lane);
        acc = wmma16(a, load_b_sw(w_lin1_0, 16, kt, task, lane), acc);
      }
      int n = task * 16 + (lane & 15);
      int mbase = (lane >> 4) * 8;
#pragma unroll
      for (int j = 0; j < 8; ++j) t_sL[(mbase + j) * 256 + n] = acc[j];
    } else {
      int t = task - 16, comp = t >> 3, ntile = t & 7;
      for (int kt = 0; kt < 8; ++kt) {
        v16h a = load_a_g_nc(msg_v + (size_t)nb * 768 + comp, 768, 3, kt, lane);
        acc = wmma16(a, load_b_sw(w_lin1_1, 8, kt, ntile, lane), acc);
      }
      int n = ntile * 16 + (lane & 15);
      int mbase = (lane >> 4) * 8;
#pragma unroll
      for (int j = 0; j < 8; ++j) t_vL[(mbase + j) * 384 + n * 3 + comp] = acc[j];
    }
  }
  __syncthreads();

  float alphav = alpha_p[0], betav = beta_p[0];
  for (int idx = tid; idx < 2048; idx += 256) {
    int m = idx >> 7, u = idx & 127;
    size_t node = (size_t)(nb + m);
    float inv = 1.0f / (density[node] * betav + alphav * 20.0f);
    float ms1 = t_sL[m * 256 + u] * inv + res_s[node * 256 + u];
    float ms2 = t_sL[m * 256 + 128 + u] * inv + res_s[node * 256 + 128 + u];
    float gate = sigmf(ms2);
    scalL[a_sw_idx(m, u)] = (_Float16)siluf(ms1);
#pragma unroll
    for (int i = 0; i < 3; ++i) {
      float mv = t_vL[m * 384 + u * 3 + i] * inv + res_v[node * 384 + u * 3 + i];
      vecL[i][a_sw_idx(m, u)] = (_Float16)(mv * gate);
    }
  }
  __syncthreads();

  // lin2: scal @ W_lin2_0 (8 tiles) and vec_i @ W_lin2_1 (24 tiles); K=128
  for (int task = wave; task < 32; task += 8) {
    const _Float16* A;
    const _Float16* B;
    int ntile, c;
    if (task < 8) { A = scalL; B = w_lin2_0; ntile = task; c = 0; }
    else { int t = task - 8; int comp = t >> 3; ntile = t & 7; A = vecL[comp]; B = w_lin2_1; c = 1 + comp; }
    v8f acc = {};
    for (int kt = 0; kt < 4; ++kt)
      acc = wmma16(load_a_sw(A, kt, lane), load_b_sw(B, 8, kt, ntile, lane), acc);
    int n = ntile * 16 + (lane & 15);
    int mbase = (lane >> 4) * 8;
#pragma unroll
    for (int j = 0; j < 8; ++j)
      out_msg[(size_t)(nb + mbase + j) * 512 + n * 4 + c] = acc[j];
  }
}

// ---------------------------------------------------------------------------
extern "C" void kernel_launch(void* const* d_in, const int* in_sizes, int n_in,
                              void* d_out, int out_size, void* d_ws, size_t ws_size,
                              hipStream_t stream) {
  const float* node_attrs = (const float*)d_in[0];
  const float* node_feats = (const float*)d_in[1];
  const float* edge_attrs = (const float*)d_in[2];
  const float* edge_feats = (const float*)d_in[3];
  const float* W_src    = (const float*)d_in[4];
  const float* W_tgt    = (const float*)d_in[5];
  const float* W_up0    = (const float*)d_in[6];
  const float* W_up1    = (const float*)d_in[7];
  const float* W_skip0  = (const float*)d_in[8];
  const float* W_skip1  = (const float*)d_in[9];
  const float* W_res0   = (const float*)d_in[10];
  const float* W_res1   = (const float*)d_in[11];
  const float* W_lin1_0 = (const float*)d_in[12];
  const float* W_lin1_1 = (const float*)d_in[13];
  const float* W_lin2_0 = (const float*)d_in[14];
  const float* W_lin2_1 = (const float*)d_in[15];
  const float* W_rad1   = (const float*)d_in[16];
  const float* W_rad2   = (const float*)d_in[17];
  const float* W_rad3   = (const float*)d_in[18];
  const float* W_rad4   = (const float*)d_in[19];
  const float* W_den1   = (const float*)d_in[20];
  const float* W_den2   = (const float*)d_in[21];
  const float* alpha    = (const float*)d_in[22];
  const float* beta     = (const float*)d_in[23];
  const int*   edge_index = (const int*)d_in[24];

  const int N = in_sizes[1] / 512;   // 10000
  const int E = in_sizes[3] / 8;     // 160000

  // ---- workspace carve-out (256B aligned slices) ----
  char*  base = (char*)d_ws;
  size_t off  = 0;
  auto af = [&](size_t n) -> float* {
    void* p = base + off; off = (off + n * sizeof(float) + 255) & ~(size_t)255;
    return (float*)p;
  };
  auto ah = [&](size_t n) -> _Float16* {
    void* p = base + off; off = (off + n * sizeof(_Float16) + 255) & ~(size_t)255;
    return (_Float16*)p;
  };

  float* up_s    = af((size_t)N * 128);
  float* up_v    = af((size_t)N * 384);
  float* res_s   = af((size_t)N * 256);
  float* res_v   = af((size_t)N * 384);
  float* msg_s   = af((size_t)N * 256);
  float* msg_v   = af((size_t)N * 768);
  float* density = af((size_t)N);
  _Float16* src_emb16 = ah((size_t)N * 128);
  _Float16* tgt_emb16 = ah((size_t)N * 128);
  _Float16* w_skip0 = ah(128 * 128);
  _Float16* w_up0   = ah(128 * 128);
  _Float16* w_res0  = ah(128 * 256);
  _Float16* w_skip1 = ah(128 * 128);
  _Float16* w_up1   = ah(128 * 128);
  _Float16* w_res1  = ah(128 * 128);
  _Float16* w_srcp  = ah(32 * 128);
  _Float16* w_tgtp  = ah(32 * 128);
  _Float16* w_rad1p = ah(288 * 64);
  _Float16* w_den1p = ah(288 * 64);
  _Float16* w_rad2  = ah(64 * 64);
  _Float16* w_rad3  = ah(64 * 64);
  _Float16* w_rad4  = ah(64 * 512);
  _Float16* w_lin1_0 = ah(256 * 256);
  _Float16* w_lin1_1 = ah(256 * 128);
  _Float16* w_lin2_0 = ah(128 * 128);
  _Float16* w_lin2_1 = ah(128 * 128);
  (void)ws_size;

  // zero scatter accumulators (graph-capture-safe)
  hipMemsetAsync(msg_s,   0, (size_t)N * 256 * sizeof(float), stream);
  hipMemsetAsync(msg_v,   0, (size_t)N * 768 * sizeof(float), stream);
  hipMemsetAsync(density, 0, (size_t)N * sizeof(float), stream);

  auto swz = [&](const float* src, _Float16* dst, int Ksrc, int Nw, int ktiles, int mode) {
    int total = ktiles * (Nw / 16) * 512;
    k_swizzleB<<<(total + 255) / 256, 256, 0, stream>>>(src, dst, Ksrc, Nw, ktiles, mode);
  };
  swz(W_skip0, w_skip0, 128, 128, 4, 0);
  swz(W_up0,   w_up0,   128, 128, 4, 0);
  swz(W_res0,  w_res0,  128, 256, 4, 0);
  swz(W_skip1, w_skip1, 128, 128, 4, 0);
  swz(W_up1,   w_up1,   128, 128, 4, 0);
  swz(W_res1,  w_res1,  128, 128, 4, 0);
  swz(W_src,   w_srcp,  10,  128, 1, 0);
  swz(W_tgt,   w_tgtp,  10,  128, 1, 0);
  swz(W_rad1,  w_rad1p, 264, 64,  9, 1);
  swz(W_den1,  w_den1p, 264, 64,  9, 1);
  swz(W_rad2,  w_rad2,  64,  64,  2, 0);
  swz(W_rad3,  w_rad3,  64,  64,  2, 0);
  swz(W_rad4,  w_rad4,  64,  512, 2, 0);
  swz(W_lin1_0, w_lin1_0, 256, 256, 8, 0);
  swz(W_lin1_1, w_lin1_1, 256, 128, 8, 0);
  swz(W_lin2_0, w_lin2_0, 128, 128, 4, 0);
  swz(W_lin2_1, w_lin2_1, 128, 128, 4, 0);

  float* out_msg = (float*)d_out;                   // (N,128,4) flat
  float* out_sc  = (float*)d_out + (size_t)N * 512; // (N,512) flat

  k_node_pre<<<N / 16, 256, 0, stream>>>(
      node_feats, node_attrs, w_skip0, w_up0, w_res0, w_skip1, w_up1, w_res1,
      w_srcp, w_tgtp, out_sc, up_s, up_v, res_s, res_v, src_emb16, tgt_emb16, N);

  k_edge<<<E / 16, 128, 0, stream>>>(
      edge_feats, edge_attrs, edge_index, src_emb16, tgt_emb16, w_rad1p, w_rad2,
      w_rad3, w_rad4, w_den1p, W_den2, up_s, up_v, msg_s, msg_v, density, E);

  k_node_post<<<N / 16, 256, 0, stream>>>(
      msg_s, msg_v, density, res_s, res_v, w_lin1_0, w_lin1_1, w_lin2_0,
      w_lin2_1, alpha, beta, out_msg, N);
}
